// GIN_5652176962226
// MI455X (gfx1250) — compile-verified, matrix operand order checked by reading, not run on hardware
//
#include <hip/hip_runtime.h>

typedef __attribute__((ext_vector_type(16))) _Float16 v16h;
typedef __attribute__((ext_vector_type(8)))  float    v8f;

// ---------------------------------------------------------------------------
// Edge scatter-add: one wave (32 lanes) per edge, 128 channels = 4 floats/lane.
// out[dst] += feat[src].  out must be pre-initialized with the self term.
// unsafeAtomicAdd lowers to global_atomic_add_f32 (no CAS loop).
// ---------------------------------------------------------------------------
__global__ void __launch_bounds__(256)
edge_scatter128(const float* __restrict__ feat,
                const int*  __restrict__ src,
                const int*  __restrict__ dst,
                float* __restrict__ out, int E)
{
    const int wave = threadIdx.x >> 5;
    const int lane = threadIdx.x & 31;
    const int e = blockIdx.x * 8 + wave;
    if (e >= E) return;
    const int s = src[e];
    const int d = dst[e];
    const float4 v = *(const float4*)(feat + (size_t)s * 128 + lane * 4);
    float* o = out + (size_t)d * 128 + lane * 4;
    unsafeAtomicAdd(o + 0, v.x);
    unsafeAtomicAdd(o + 1, v.y);
    unsafeAtomicAdd(o + 2, v.z);
    unsafeAtomicAdd(o + 3, v.w);
}

// ---------------------------------------------------------------------------
// Zero-pad the 256x40 head weights to 256x48 (and bias to 48) so the GEMM
// inner loop never needs a bounds check.
// ---------------------------------------------------------------------------
__global__ void __launch_bounds__(256)
pad_head(const float* __restrict__ Wl, const float* __restrict__ bl,
         float* __restrict__ Wp, float* __restrict__ bp)
{
    const int i = blockIdx.x * 256 + threadIdx.x;
    if (i < 256 * 48) {
        const int r = i / 48, c = i % 48;
        Wp[i] = (c < 40) ? Wl[r * 40 + c] : 0.f;
    }
    if (i < 48) bp[i] = (i < 40) ? bl[i] : 0.f;
}

// ---------------------------------------------------------------------------
// Fused GEMM + bias + optional ReLU:  C[M,Nout] = act(A[M,K] * W[K,NW] + b)
// K and NW (padded W width) are compile-time so the MAC loop fully unrolls:
// straight-line global_load_b64 / b32 with immediate offsets feeding a
// back-to-back v_wmma_f32_16x16x32_f16 chain (f32 accumulation).
//
// One wave owns a 16-column strip: it loads the full K-panel of B fragments
// once (register resident) and sweeps MT=4 M-tiles, reusing B.
//
// Fragment layouts per ISA 7.12.2 (16-bit, wave32):
//   A 16x32: lane L -> row m=L%16, half h=L/16; half-pair v holds
//            K = (v<4 ? 2v : 16+2(v-4)) + 8h.
//   B 32x16: mirrored with lane -> column n.
//   C/D 16x16: lane L -> column n=L%16, rows m = 8*(L/16)+r in VGPR r.
//
// Requires M % 16 == 0 (holds: 50000 = 3125*16).
// ---------------------------------------------------------------------------
template<int K, int NW, bool RELU>
__global__ void __launch_bounds__(256)
gemm_bias_act(const float* __restrict__ A, const float* __restrict__ W,
              const float* __restrict__ bias, float* __restrict__ C,
              int Nout, int mtiles, int mgroups)
{
    constexpr int MT = 4;        // M-tiles swept per wave
    constexpr int NT = NW / 16;  // column tiles
    constexpr int KC = K / 32;   // K chunks

    const int lane = threadIdx.x & 31;
    const int wave = threadIdx.x >> 5;
    const int tile = blockIdx.x * 8 + wave;
    if (tile >= mgroups * NT) return;
    const int nt = tile % NT;
    const int mg = tile / NT;

    const int lm = lane & 15;    // m for A-frag, n for B/C-frag
    const int lh = lane >> 4;    // lane-half selects K sub-range / row group
    const int col = nt * 16 + lm;

    // Register-resident B panel for this wave's 16-column strip.
    v16h b[KC];
    const float* Wcol = W + col;
#pragma unroll
    for (int c = 0; c < KC; ++c) {
#pragma unroll
        for (int v = 0; v < 8; ++v) {
            const int kk = c * 32 + ((v < 4) ? (2 * v) : (16 + 2 * (v - 4))) + 8 * lh;
            b[c][2 * v]     = (_Float16)Wcol[(size_t)kk * NW];
            b[c][2 * v + 1] = (_Float16)Wcol[(size_t)(kk + 1) * NW];
        }
    }
    const float bv = bias[col];

    const int mt0 = mg * MT;
    const int mt1 = (mt0 + MT < mtiles) ? (mt0 + MT) : mtiles;
    for (int mt = mt0; mt < mt1; ++mt) {
        const float* Arow = A + (size_t)(mt * 16 + lm) * K;
        v8f acc = {};
#pragma unroll
        for (int c = 0; c < KC; ++c) {
            v16h a;
#pragma unroll
            for (int v = 0; v < 8; ++v) {
                const int kk = c * 32 + ((v < 4) ? (2 * v) : (16 + 2 * (v - 4))) + 8 * lh;
                const float2 av = *(const float2*)(Arow + kk);
                a[2 * v]     = (_Float16)av.x;
                a[2 * v + 1] = (_Float16)av.y;
            }
            acc = __builtin_amdgcn_wmma_f32_16x16x32_f16(
                false, a, false, b[c], (short)0, acc, false, false);
        }
        // Store (guard only matters for the padded 40-wide head).
        if (col < Nout) {
            float* Crow = C + (size_t)(mt * 16 + 8 * lh) * Nout + col;
#pragma unroll
            for (int r = 0; r < 8; ++r) {
                float o = acc[r] + bv;
                if (RELU) o = fmaxf(o, 0.f);
                Crow[(size_t)r * Nout] = o;
            }
        }
    }
}

template<int K, int NW, bool RELU>
static inline void launch_gemm(const float* A, const float* W, const float* b,
                               float* C, int M, int Nout, hipStream_t stream)
{
    const int mtiles  = M / 16;            // M % 16 == 0
    const int mgroups = (mtiles + 3) / 4;
    const int tiles   = mgroups * (NW / 16);
    const int blocks  = (tiles + 7) / 8;   // 8 waves per 256-thread block
    gemm_bias_act<K, NW, RELU><<<blocks, 256, 0, stream>>>(A, W, b, C, Nout, mtiles, mgroups);
}

extern "C" void kernel_launch(void* const* d_in, const int* in_sizes, int n_in,
                              void* d_out, int out_size, void* d_ws, size_t ws_size,
                              hipStream_t stream)
{
    const float* x  = (const float*)d_in[0];
    const int*   ei = (const int*)  d_in[1];   // harness contract: integer -> int32
    const float* W1 = (const float*)d_in[2];
    const float* b1 = (const float*)d_in[3];
    const float* W2 = (const float*)d_in[4];
    const float* b2 = (const float*)d_in[5];
    const float* W3 = (const float*)d_in[6];
    const float* b3 = (const float*)d_in[7];
    const float* W4 = (const float*)d_in[8];
    const float* b4 = (const float*)d_in[9];
    const float* Wl = (const float*)d_in[10];
    const float* bl = (const float*)d_in[11];
    float* out = (float*)d_out;

    const int N = in_sizes[0] / 128;
    const int E = in_sizes[1] / 2;
    const int* src = ei;
    const int* dst = ei + E;

    // Workspace layout (floats):
    //   bufA = [0,      N*128) : z1, then h1b (post-MLP1)
    //   bufB = [N*128,  N*256) : h1, then z2
    //   bufC = [N*256,  N*512) : h3 (256-d)
    //   bufD = [0,      N*256) : h4 (256-d, reuses bufA+bufB)
    //   Wp   = [N*512, +256*48), bp = [.. +48) : zero-padded head
    float* ws   = (float*)d_ws;
    float* bufA = ws;
    float* bufB = ws + (size_t)N * 128;
    float* bufC = ws + (size_t)N * 256;
    float* bufD = ws;
    float* Wp   = ws + (size_t)N * 512;
    float* bp   = Wp + 256 * 48;

    const size_t feat128 = (size_t)N * 128 * sizeof(float);
    const int scatterBlocks = (E + 7) / 8;

    // Pad the head weights early (independent of the rest of the pipeline).
    pad_head<<<(256 * 48 + 255) / 256, 256, 0, stream>>>(Wl, bl, Wp, bp);

    // ---- Layer 1: z1 = x + scatter(x) -------------------------------------
    hipMemcpyAsync(bufA, x, feat128, hipMemcpyDeviceToDevice, stream);
    edge_scatter128<<<scatterBlocks, 256, 0, stream>>>(x, src, dst, bufA, E);

    // h1  = relu(z1 @ W1 + b1)   [N,128] -> bufB
    launch_gemm<128, 128, true>(bufA, W1, b1, bufB, N, 128, stream);
    // h1b = relu(h1 @ W2 + b2)   [N,128] -> bufA
    launch_gemm<128, 128, true>(bufB, W2, b2, bufA, N, 128, stream);

    // ---- Layer 2: z2 = h1b + scatter(h1b) ---------------------------------
    hipMemcpyAsync(bufB, bufA, feat128, hipMemcpyDeviceToDevice, stream);
    edge_scatter128<<<scatterBlocks, 256, 0, stream>>>(bufA, src, dst, bufB, E);

    // h3 = relu(z2 @ W3 + b3)    [N,256] -> bufC
    launch_gemm<128, 256, true>(bufB, W3, b3, bufC, N, 256, stream);
    // h4 = relu(h3 @ W4 + b4)    [N,256] -> bufD (reuses bufA+bufB)
    launch_gemm<256, 256, true>(bufC, W4, b4, bufD, N, 256, stream);

    // ---- Output head: out = h4 @ Wp + bp    [N,40] (padded to 48) ---------
    launch_gemm<256, 48, false>(bufD, Wp, bp, out, N, 40, stream);
}
